// Img_Graph_Multi_Graph_Net_17995912970743
// MI455X (gfx1250) — matrix operand order repaired and checked by reading, not compiled, not run on hardware
//
#include <hip/hip_runtime.h>
#include <hip/hip_bf16.h>

typedef __attribute__((ext_vector_type(16))) __bf16 v16bf;
typedef __attribute__((ext_vector_type(8)))  float  v8f;
typedef int v4i __attribute__((vector_size(16)));   // matches builtin's V4i param

#define WMMA_BF16(a, b, c) \
  __builtin_amdgcn_wmma_f32_16x16x32_bf16(false, (a), false, (b), (short)0, (c), false, false)

// ---------------------------------------------------------------------------
// CDNA5 async global->LDS copy (GLOBAL_LOAD_ASYNC_TO_LDS_B128, ASYNCcnt),
// guarded so the kernel still compiles (plain-copy fallback) if the builtin
// is not exposed by this toolchain.
// ---------------------------------------------------------------------------
#if defined(__HIP_DEVICE_COMPILE__) && __has_builtin(__builtin_amdgcn_global_load_async_to_lds_b128)
#define USE_ASYNC 1
#else
#define USE_ASYNC 0
#endif

#if USE_ASYNC
#if __has_builtin(__builtin_amdgcn_s_wait_asynccnt)
#define ASYNC_WAIT(n) __builtin_amdgcn_s_wait_asynccnt(n)
#else
#define ASYNC_WAIT(n) asm volatile("s_wait_asynccnt %0" ::"n"(n))
#endif
#else
#define ASYNC_WAIT(n) ((void)0)
#endif

__device__ __forceinline__ void copy16_g2l(const __bf16* g, __bf16* l) {
#if USE_ASYNC
  __builtin_amdgcn_global_load_async_to_lds_b128(
      (__attribute__((address_space(1))) v4i*)(v4i*)(void*)g,
      (__attribute__((address_space(3))) v4i*)(v4i*)(void*)l, 0, 0);
#else
  *(uint4*)(void*)l = *(const uint4*)(const void*)g;
#endif
}

// ---------------------------------------------------------------------------
// Fragment loaders (CDNA5 ISA 7.12.2 wave32 layouts, 16-bit data)
// ---------------------------------------------------------------------------
__device__ __forceinline__ v16bf ld_afrag(const __bf16* base, int ld, int kbase, int lane) {
  int row  = lane & 15;
  int koff = kbase + ((lane & 16) ? 8 : 0);
  union { v16bf v; unsigned u[8]; } f;
  const unsigned* p = (const unsigned*)(base + row * ld);
#pragma unroll
  for (int j = 0; j < 8; ++j) {
    int k = 2 * j + (j >= 4 ? 8 : 0) + koff;  // always even
    f.u[j] = p[k >> 1];
  }
  return f.v;
}

__device__ __forceinline__ v16bf ld_bfrag(const __bf16* base, int ld, int kbase, int lane) {
  int n    = lane & 15;
  int koff = kbase + ((lane & 16) ? 8 : 0);
  union { v16bf v; unsigned u[8]; } f;
  const unsigned* p = (const unsigned*)(base + n * ld);
#pragma unroll
  for (int j = 0; j < 8; ++j) f.u[j] = p[(koff >> 1) + j + (j >= 4 ? 4 : 0)];
  return f.v;
}

// ---------------------------------------------------------------------------
// f32 -> bf16 conversion (grid-stride)
// ---------------------------------------------------------------------------
__global__ void f32_to_bf16_kernel(const float* __restrict__ in, __bf16* __restrict__ out, int n) {
  for (int i = blockIdx.x * blockDim.x + threadIdx.x; i < n; i += gridDim.x * blockDim.x)
    out[i] = (__bf16)in[i];
}

// ---------------------------------------------------------------------------
// GEMM: C[M,N] = A[M,K] @ B[N,K]^T + bias. A,B bf16; accumulate fp32.
// Double-buffered LDS, async global->LDS prefetch of the next K-tile.
// Block: 128 threads (4 waves), tile 64(M) x 64(N), K-step 32.
// mode: 0 = none, 1 = leaky relu, 2 = sanitize (nan->0, clip +-1000)
// ---------------------------------------------------------------------------
__global__ __launch_bounds__(128) void gemm_bf16_kernel(
    const __bf16* __restrict__ A, const __bf16* __restrict__ B,
    const float* __restrict__ bias, float* __restrict__ Cf, __bf16* __restrict__ Cb,
    int M, int N, int K, int mode) {
  __shared__ __bf16 sa[2][64 * 32];
  __shared__ __bf16 sb[2][64 * 32];
  const int tid = threadIdx.x, lane = tid & 31, w = tid >> 5;
  const int rowBase = blockIdx.x * 64, colBase = blockIdx.y * 64;

  // 16B chunks: 256 for A-tile + 256 for B-tile -> 4 per thread per K-step
  auto copy_tile = [&](int buf, int kb) {
#pragma unroll
    for (int j = 0; j < 4; ++j) {
      int c    = tid + j * 128;
      int half = c >> 8;        // 0 = A, 1 = B
      int cc   = c & 255;
      int r    = cc >> 2;       // tile row 0..63
      int co   = (cc & 3) * 8;  // element offset in row
      const __bf16* g = (half ? B : A) + (size_t)((half ? colBase : rowBase) + r) * K + kb + co;
      __bf16* l = (half ? sb[buf] : sa[buf]) + r * 32 + co;
      copy16_g2l(g, l);
    }
  };

  v8f zero = {0.f, 0.f, 0.f, 0.f, 0.f, 0.f, 0.f, 0.f};
  v8f acc[4] = {zero, zero, zero, zero};

  copy_tile(0, 0);
  int buf = 0;
  for (int kb = 0; kb < K; kb += 32) {
    if (kb + 32 < K) {
      copy_tile(buf ^ 1, kb + 32);  // prefetch next tile
      ASYNC_WAIT(4);                // retire current tile, keep next in flight
    } else {
      ASYNC_WAIT(0);
    }
    __syncthreads();
    v16bf a = ld_afrag(sa[buf] + w * 16 * 32, 32, 0, lane);
#pragma unroll
    for (int t = 0; t < 4; ++t) {
      v16bf b = ld_bfrag(sb[buf] + t * 16 * 32, 32, 0, lane);
      acc[t] = WMMA_BF16(a, b, acc[t]);
    }
    __syncthreads();
    buf ^= 1;
  }

  const int row0 = rowBase + w * 16 + ((lane & 16) ? 8 : 0);
#pragma unroll
  for (int t = 0; t < 4; ++t) {
    int col  = colBase + t * 16 + (lane & 15);
    float bv = bias ? bias[col] : 0.f;
#pragma unroll
    for (int r = 0; r < 8; ++r) {
      float v = acc[t][r] + bv;
      if (mode == 1) v = v > 0.f ? v : 0.2f * v;
      if (mode == 2) {
        if (v != v) v = 0.f;
        v = fminf(fmaxf(v, -1000.f), 1000.f);
      }
      size_t idx = (size_t)(row0 + r) * N + col;
      if (Cf) Cf[idx] = v;
      if (Cb) Cb[idx] = (__bf16)v;
    }
  }
}

// ---------------------------------------------------------------------------
// Flash-style causal MHA on bf16 qkv (N,768) = [q|k|v]; head slice d=0..63.
// Grid: (N/64, H). Block: 128 threads (4 waves). Output bf16 (N,256).
// ---------------------------------------------------------------------------
__global__ __launch_bounds__(128) void mha_flash_kernel(
    const __bf16* __restrict__ qkv, __bf16* __restrict__ attn_out) {
  __shared__ __bf16 qs[64 * 64];   // [row][d]
  __shared__ __bf16 ks[64 * 64];   // [key][d]
  __shared__ __bf16 vst[64 * 64];  // [d][key]  (transposed V)
  __shared__ __bf16 ps[64 * 64];   // [row][key] probabilities
  __shared__ float  ss[64 * 64];   // [row][key] raw scores
  __shared__ float  mrow[64], lrow[64], arow[64];

  const int tid = threadIdx.x, lane = tid & 31, w = tid >> 5;
  const int h = blockIdx.y;
  const int qbase = blockIdx.x * 64;
  const int mloc = w * 16 + ((lane & 16) ? 8 : 0);

  // Q tile: 512 x 16B chunks, async
#pragma unroll
  for (int j = 0; j < 4; ++j) {
    int c = tid + j * 128;
    int r = c >> 3, co = (c & 7) * 8;
    copy16_g2l(qkv + (size_t)(qbase + r) * 768 + h * 64 + co, qs + r * 64 + co);
  }
  if (tid < 64) { mrow[tid] = -1e30f; lrow[tid] = 0.f; }
  ASYNC_WAIT(0);
  __syncthreads();

  v16bf aq0 = ld_afrag(qs + w * 16 * 64, 64, 0, lane);
  v16bf aq1 = ld_afrag(qs + w * 16 * 64, 64, 32, lane);

  v8f zero = {0.f, 0.f, 0.f, 0.f, 0.f, 0.f, 0.f, 0.f};
  v8f o[4] = {zero, zero, zero, zero};

  for (int jb = 0; jb <= (int)blockIdx.x; ++jb) {
    const int kbase = jb * 64;
    const bool diag = (jb == (int)blockIdx.x);
    __syncthreads();  // previous reads of ks/vst complete
    // K tile: async copy (natural layout); V tile: transposed store
#pragma unroll
    for (int j = 0; j < 4; ++j) {
      int c = tid + j * 128;
      int r = c >> 3, co = (c & 7) * 8;
      copy16_g2l(qkv + (size_t)(kbase + r) * 768 + 256 + h * 64 + co, ks + r * 64 + co);
    }
    for (int i = tid; i < 4096; i += 128) {
      int r = i >> 6, d = i & 63;
      vst[d * 64 + r] = qkv[(size_t)(kbase + r) * 768 + 512 + h * 64 + d];
    }
    ASYNC_WAIT(0);
    __syncthreads();

    // S = (Q K^T)/8, causal mask on diagonal block
#pragma unroll
    for (int t = 0; t < 4; ++t) {
      v8f s = zero;
      s = WMMA_BF16(aq0, ld_bfrag(ks + t * 16 * 64, 64, 0, lane), s);
      s = WMMA_BF16(aq1, ld_bfrag(ks + t * 16 * 64, 64, 32, lane), s);
      int colL = t * 16 + (lane & 15);
#pragma unroll
      for (int r = 0; r < 8; ++r) {
        float v = s[r] * 0.125f;
        if (diag && (kbase + colL) > (qbase + mloc + r)) v = -1e30f;
        ss[(mloc + r) * 64 + colL] = v;
      }
    }
    __syncthreads();

    // online softmax: one thread per query row
    if (tid < 64) {
      float mo = mrow[tid], mx = mo;
      for (int c = 0; c < 64; ++c) mx = fmaxf(mx, ss[tid * 64 + c]);
      float alpha = __expf(mo - mx);
      float sum = 0.f;
      for (int c = 0; c < 64; ++c) {
        float p = __expf(ss[tid * 64 + c] - mx);
        sum += p;
        ps[tid * 64 + c] = (__bf16)p;
      }
      mrow[tid] = mx;
      lrow[tid] = lrow[tid] * alpha + sum;
      arow[tid] = alpha;
    }
    __syncthreads();

    // O = O*alpha + P @ V
    v16bf ap0 = ld_afrag(ps + w * 16 * 64, 64, 0, lane);
    v16bf ap1 = ld_afrag(ps + w * 16 * 64, 64, 32, lane);
    float al[8];
#pragma unroll
    for (int r = 0; r < 8; ++r) al[r] = arow[mloc + r];
#pragma unroll
    for (int t = 0; t < 4; ++t) {
#pragma unroll
      for (int r = 0; r < 8; ++r) o[t][r] *= al[r];
      o[t] = WMMA_BF16(ap0, ld_bfrag(vst + t * 16 * 64, 64, 0, lane), o[t]);
      o[t] = WMMA_BF16(ap1, ld_bfrag(vst + t * 16 * 64, 64, 32, lane), o[t]);
    }
  }
  __syncthreads();

#pragma unroll
  for (int t = 0; t < 4; ++t) {
    int d = t * 16 + (lane & 15);
#pragma unroll
    for (int r = 0; r < 8; ++r) {
      float v = o[t][r] / lrow[mloc + r];
      attn_out[(size_t)(qbase + mloc + r) * 256 + h * 64 + d] = (__bf16)v;
    }
  }
}

// ---------------------------------------------------------------------------
// Edge attention kernels (fp32, atomic segment reductions)
// ---------------------------------------------------------------------------
__global__ void zero_stats_kernel(unsigned* __restrict__ amax_bits,
                                  float* __restrict__ denom, int n) {
  int i = blockIdx.x * blockDim.x + threadIdx.x;
  if (i >= n) return;
  amax_bits[i] = 0u;
  denom[i] = 0.f;
}

__device__ __forceinline__ unsigned f32_order_enc(float f) {
  unsigned b = __float_as_uint(f);
  return (b & 0x80000000u) ? ~b : (b | 0x80000000u);
}
__device__ __forceinline__ float f32_order_dec(unsigned e) {
  unsigned b = (e & 0x80000000u) ? (e & 0x7FFFFFFFu) : ~e;
  return __uint_as_float(b);
}

__global__ void edge_logits_kernel(const float* __restrict__ q, const float* __restrict__ k,
                                   const int* __restrict__ src, const int* __restrict__ dst,
                                   float* __restrict__ logits, unsigned* __restrict__ amax_bits,
                                   int E) {
  int idx = blockIdx.x * blockDim.x + threadIdx.x;
  if (idx >= E * 4) return;
  int e = idx >> 2, hh = idx & 3;
  int s = src[e], d = dst[e];
  const float4* qp = (const float4*)(q + (size_t)d * 256 + hh * 64);
  const float4* kp = (const float4*)(k + (size_t)s * 256 + hh * 64);
  float acc = 0.f;
#pragma unroll
  for (int i = 0; i < 16; ++i) {
    float4 a = qp[i], b = kp[i];
    acc += a.x * b.x + a.y * b.y + a.z * b.z + a.w * b.w;
  }
  acc *= 0.125f;
  logits[idx] = acc;
  atomicMax(&amax_bits[d * 4 + hh], f32_order_enc(acc));
}

__global__ void edge_exp_kernel(const float* __restrict__ logits,
                                const int* __restrict__ dst,
                                const unsigned* __restrict__ amax_bits,
                                float* __restrict__ exbuf, float* __restrict__ denom, int E) {
  int idx = blockIdx.x * blockDim.x + threadIdx.x;
  if (idx >= E * 4) return;
  int e = idx >> 2, hh = idx & 3;
  int d = dst[e];
  unsigned enc = amax_bits[d * 4 + hh];
  float amax = (enc == 0u) ? 0.f : f32_order_dec(enc);
  if (!(amax >= -3e38f && amax <= 3e38f)) amax = 0.f;
  float ex = __expf(logits[idx] - amax);
  exbuf[idx] = ex;
  atomicAdd(&denom[d * 4 + hh], ex);
}

__global__ void edge_agg_kernel(const float* __restrict__ exbuf,
                                const float* __restrict__ denom,
                                const float* __restrict__ v,
                                const int* __restrict__ src, const int* __restrict__ dst,
                                float* __restrict__ out, int E) {
  int idx = blockIdx.x * blockDim.x + threadIdx.x;
  if (idx >= E * 4) return;
  int e = idx >> 2, hh = idx & 3;
  int s = src[e], d = dst[e];
  float wgt = exbuf[idx] / (denom[d * 4 + hh] + 1e-16f);
  const float* vp = v + (size_t)s * 256 + hh * 64;
  float* op = out + (size_t)d * 256 + hh * 64;
  for (int i = 0; i < 64; ++i) atomicAdd(&op[i], wgt * vp[i]);
}

// per-channel instance norm over 4096 nodes + lReLU + sanitize -> bf16 f[:, colOff+ch]
__global__ __launch_bounds__(256) void instnorm_kernel(const float* __restrict__ in,
                                                       __bf16* __restrict__ out, int colOff) {
  __shared__ float ssum[256], ssq[256];
  int ch = blockIdx.x;
  float s = 0.f, sq = 0.f;
  for (int r = threadIdx.x; r < 4096; r += 256) {
    float v = in[(size_t)r * 256 + ch];
    s += v; sq += v * v;
  }
  ssum[threadIdx.x] = s; ssq[threadIdx.x] = sq;
  __syncthreads();
  for (int off = 128; off > 0; off >>= 1) {
    if (threadIdx.x < off) {
      ssum[threadIdx.x] += ssum[threadIdx.x + off];
      ssq[threadIdx.x]  += ssq[threadIdx.x + off];
    }
    __syncthreads();
  }
  float mean = ssum[0] * (1.f / 4096.f);
  float var  = ssq[0] * (1.f / 4096.f) - mean * mean;
  float rstd = rsqrtf(var + 1e-5f);
  for (int r = threadIdx.x; r < 4096; r += 256) {
    float v = (in[(size_t)r * 256 + ch] - mean) * rstd;
    v = v > 0.f ? v : 0.2f * v;
    if (v != v) v = 0.f;
    v = fminf(fmaxf(v, -1000.f), 1000.f);
    out[(size_t)r * 512 + colOff + ch] = (__bf16)v;
  }
}

// final head: logits = c1 @ W_c2^T + b_c2 ; probs = softmax.
__global__ void head_kernel(const float* __restrict__ c1, const float* __restrict__ W2,
                            const float* __restrict__ b2, float* __restrict__ out) {
  int n = blockIdx.x * blockDim.x + threadIdx.x;
  if (n >= 4096) return;
  const float* f = c1 + (size_t)n * 128;
  float l0 = b2[0], l1 = b2[1];
  for (int i = 0; i < 128; ++i) {
    float v = f[i];
    l0 += v * W2[i];
    l1 += v * W2[128 + i];
  }
  out[n * 2] = l0;
  out[n * 2 + 1] = l1;
  float mx = fmaxf(l0, l1);
  float e0 = __expf(l0 - mx), e1 = __expf(l1 - mx);
  float inv = 1.f / (e0 + e1);
  out[8192 + n * 2] = e0 * inv;
  out[8192 + n * 2 + 1] = e1 * inv;
}

// ---------------------------------------------------------------------------
// Host launch
// ---------------------------------------------------------------------------
extern "C" void kernel_launch(void* const* d_in, const int* in_sizes, int n_in,
                              void* d_out, int out_size, void* d_ws, size_t ws_size,
                              hipStream_t stream) {
  (void)in_sizes; (void)n_in; (void)out_size; (void)ws_size;
  const int N = 4096, E = 262144;

  const float* img_feat = (const float*)d_in[2];
  const int*   adj      = (const int*)d_in[3];
  const float* W_img = (const float*)d_in[4];  const float* b_img = (const float*)d_in[5];
  const float* W_qkv = (const float*)d_in[6];  const float* b_qkv = (const float*)d_in[7];
  const float* W_o   = (const float*)d_in[8];  const float* b_o   = (const float*)d_in[9];
  const float* Wq_o = (const float*)d_in[10]; const float* bq_o = (const float*)d_in[11];
  const float* Wk_o = (const float*)d_in[12]; const float* bk_o = (const float*)d_in[13];
  const float* Wv_o = (const float*)d_in[14]; const float* bv_o = (const float*)d_in[15];
  const float* Ws_o = (const float*)d_in[16]; const float* bs_o = (const float*)d_in[17];
  const float* Wq_a = (const float*)d_in[18]; const float* bq_a = (const float*)d_in[19];
  const float* Wk_a = (const float*)d_in[20]; const float* bk_a = (const float*)d_in[21];
  const float* Wv_a = (const float*)d_in[22]; const float* bv_a = (const float*)d_in[23];
  const float* Ws_a = (const float*)d_in[24]; const float* bs_a = (const float*)d_in[25];
  const float* W_c1 = (const float*)d_in[26]; const float* b_c1 = (const float*)d_in[27];
  const float* W_c2 = (const float*)d_in[28]; const float* b_c2 = (const float*)d_in[29];

  const int* e_src = adj;
  const int* e_dst = adj + E;

  char* ws = (char*)d_ws;
  auto carve = [&](size_t bytes) -> void* {
    void* p = (void*)ws;
    ws += (bytes + 255) & ~(size_t)255;
    return p;
  };
  // bf16 tensors
  __bf16* img_bf   = (__bf16*)carve((size_t)N * 2048 * 2);
  __bf16* wimg_bf  = (__bf16*)carve((size_t)256 * 2048 * 2);
  __bf16* wqkv_bf  = (__bf16*)carve((size_t)768 * 256 * 2);
  __bf16* wo_bf    = (__bf16*)carve((size_t)256 * 256 * 2);
  __bf16* wconv_bf[8];
  for (int i = 0; i < 8; ++i) wconv_bf[i] = (__bf16*)carve((size_t)256 * 256 * 2);
  __bf16* wc1_bf   = (__bf16*)carve((size_t)128 * 512 * 2);
  __bf16* h_bf     = (__bf16*)carve((size_t)N * 256 * 2);
  __bf16* qkv_bf   = (__bf16*)carve((size_t)N * 768 * 2);
  __bf16* attn_bf  = (__bf16*)carve((size_t)N * 256 * 2);
  __bf16* hattn_bf = (__bf16*)carve((size_t)N * 256 * 2);
  __bf16* fcat_bf  = (__bf16*)carve((size_t)N * 512 * 2);
  // fp32 tensors
  float* qf       = (float*)carve((size_t)N * 256 * 4);
  float* kf       = (float*)carve((size_t)N * 256 * 4);
  float* vf       = (float*)carve((size_t)N * 256 * 4);
  float* convout  = (float*)carve((size_t)N * 256 * 4);
  float* logitsE  = (float*)carve((size_t)E * 4 * 4);
  float* exE      = (float*)carve((size_t)E * 4 * 4);
  unsigned* amaxB = (unsigned*)carve((size_t)N * 4 * 4);
  float* denom    = (float*)carve((size_t)N * 4 * 4);
  float* c1       = (float*)carve((size_t)N * 128 * 4);

  auto cvt = [&](const float* src, __bf16* dst, int n) {
    int blocks = (n + 255) / 256;
    if (blocks > 4096) blocks = 4096;
    f32_to_bf16_kernel<<<blocks, 256, 0, stream>>>(src, dst, n);
  };
  auto gemm = [&](const __bf16* A, const __bf16* B, const float* bias, float* Cf, __bf16* Cb,
                  int M, int Nc, int K, int mode) {
    gemm_bf16_kernel<<<dim3(M / 64, Nc / 64), 128, 0, stream>>>(A, B, bias, Cf, Cb, M, Nc, K, mode);
  };

  // 0) one-time bf16 conversions of inputs/weights
  cvt(img_feat, img_bf, N * 2048);
  cvt(W_img, wimg_bf, 256 * 2048);
  cvt(W_qkv, wqkv_bf, 768 * 256);
  cvt(W_o, wo_bf, 256 * 256);
  const float* wconv_src[8] = {Wq_o, Wk_o, Wv_o, Ws_o, Wq_a, Wk_a, Wv_a, Ws_a};
  for (int i = 0; i < 8; ++i) cvt(wconv_src[i], wconv_bf[i], 256 * 256);
  cvt(W_c1, wc1_bf, 128 * 512);

  // 1) h = img_feat @ W_img^T + b_img   (bf16 out)
  gemm(img_bf, wimg_bf, b_img, nullptr, h_bf, N, 256, 2048, 0);

  // 2) MHA: qkv -> flash attention -> out proj (+fused sanitize)
  gemm(h_bf, wqkv_bf, b_qkv, nullptr, qkv_bf, N, 768, 256, 0);
  mha_flash_kernel<<<dim3(N / 64, 4), 128, 0, stream>>>(qkv_bf, attn_bf);
  gemm(attn_bf, wo_bf, b_o, nullptr, hattn_bf, N, 256, 256, 2);

  // 3) TransformerConv x2
  auto conv = [&](const __bf16* x, const __bf16* Wq, const float* bq, const __bf16* Wk,
                  const float* bk, const __bf16* Wv, const float* bv, const __bf16* Wsk,
                  const float* bsk, int colOff) {
    gemm(x, Wq, bq, qf, nullptr, N, 256, 256, 0);
    gemm(x, Wk, bk, kf, nullptr, N, 256, 256, 0);
    gemm(x, Wv, bv, vf, nullptr, N, 256, 256, 0);
    gemm(x, Wsk, bsk, convout, nullptr, N, 256, 256, 0);  // skip branch seeds output
    zero_stats_kernel<<<(N * 4) / 256, 256, 0, stream>>>(amaxB, denom, N * 4);
    edge_logits_kernel<<<(E * 4) / 256, 256, 0, stream>>>(qf, kf, e_src, e_dst, logitsE, amaxB, E);
    edge_exp_kernel<<<(E * 4) / 256, 256, 0, stream>>>(logitsE, e_dst, amaxB, exE, denom, E);
    edge_agg_kernel<<<(E * 4) / 256, 256, 0, stream>>>(exE, denom, vf, e_src, e_dst, convout, E);
    instnorm_kernel<<<256, 256, 0, stream>>>(convout, fcat_bf, colOff);
  };
  conv(h_bf,     wconv_bf[0], bq_o, wconv_bf[1], bk_o, wconv_bf[2], bv_o, wconv_bf[3], bs_o, 0);
  conv(hattn_bf, wconv_bf[4], bq_a, wconv_bf[5], bk_a, wconv_bf[6], bv_a, wconv_bf[7], bs_a, 256);

  // 4) classifier (fused leaky relu) + head
  gemm(fcat_bf, wc1_bf, b_c1, c1, nullptr, N, 128, 512, 1);
  head_kernel<<<16, 256, 0, stream>>>(c1, W_c2, b_c2, (float*)d_out);
}
